// my_stn_2396591751251
// MI455X (gfx1250) — matrix operand order repaired and checked by reading, not compiled, not run on hardware
//
#include <hip/hip_runtime.h>
#include <cmath>
#include <cstdint>

typedef unsigned short u16;
typedef __attribute__((ext_vector_type(16))) __bf16        v16bf;
typedef __attribute__((ext_vector_type(8)))  float         v8f;
typedef __attribute__((ext_vector_type(4)))  unsigned int  u32x4;
typedef __attribute__((ext_vector_type(4)))  unsigned int  du32x4;
typedef __attribute__((ext_vector_type(8)))  int           di32x8;
typedef __attribute__((ext_vector_type(4)))  int           di32x4;

union Frag16 {            // 16 bf16 = 8 VGPRs, one WMMA A/B operand
  v16bf v;
  u32x4 q[2];
  u16   u[16];
};

__device__ __forceinline__ u16 f2bf(float f) {   // round-to-nearest-even
  unsigned u = __builtin_bit_cast(unsigned, f);
  u += 0x7FFFu + ((u >> 16) & 1u);
  return (u16)(u >> 16);
}

__device__ __forceinline__ v8f wmma_bf16(v16bf a, v16bf b, v8f c) {
  return __builtin_amdgcn_wmma_f32_16x16x32_bf16(false, a, false, b, (short)0, c,
                                                 false, false);
}

// ---- Tensor Data Mover: 2-D tile (bf16) global -> LDS, per ISA 08 §8.3/8.4 --
// D# group0: [1:0]=count=1, [63:32]=lds_addr, [120:64]=global_addr, [127:126]=2
// D# group1: [17:16]=data_size(1=2B), [79:48]=tensor_dim0, [111:80]=tensor_dim1,
//            [127:112]=tile_dim0, [143:128]=tile_dim1, [207:160]=dim0_stride
// Toolchain uses the 6-arg builtin: (g0, g1, g2, g3, g4, cpol).
__device__ __forceinline__ void tdm_load_2d(unsigned lds_off, const void* gptr,
                                            unsigned tensor_d0, unsigned tensor_d1,
                                            unsigned stride0_elems,
                                            unsigned tile_d0, unsigned tile_d1) {
  unsigned long long ga = (unsigned long long)(uintptr_t)gptr;
  du32x4 g0;
  g0[0] = 1u;                                           // count=1 (user D#)
  g0[1] = lds_off;                                      // lds_addr (bytes)
  g0[2] = (unsigned)(ga & 0xFFFFFFFFu);                 // global_addr[31:0]
  g0[3] = (unsigned)((ga >> 32) & 0x01FFFFFFu) | (2u << 30);  // [56:32] | type=2
  di32x8 g1;
  g1[0] = 0x00010000;                                   // data_size=2B, mask=0
  g1[1] = (int)((tensor_d0 & 0xFFFFu) << 16);           // tensor_dim0[15:0]
  g1[2] = (int)((tensor_d0 >> 16) | ((tensor_d1 & 0xFFFFu) << 16));
  g1[3] = (int)((tensor_d1 >> 16) | (tile_d0 << 16));   // dim1 hi | tile_dim0
  g1[4] = (int)(tile_d1 & 0xFFFFu);                     // tile_dim1 (tile_dim2=0)
  g1[5] = (int)stride0_elems;                           // dim0_stride[31:0]
  g1[6] = 0;                                            // stride hi / dim1_stride
  g1[7] = 0;
  di32x4 z4 = {0, 0, 0, 0};
  di32x8 z8 = {0, 0, 0, 0, 0, 0, 0, 0};
  __builtin_amdgcn_tensor_load_to_lds(g0, g1, z4, z4, z8, 0);
}

// ---------------------------------------------------------------- prep ------
// dst[n*KPAD + k] = (n<Nsrc && k<Ksrc) ? bf16(src[k*Nsrc+n]) : 0   (B^T, padded)
__global__ void prep_bt(const float* __restrict__ src, u16* __restrict__ dst,
                        int Nsrc, int NPAD, int Ksrc, int KPAD) {
  int total = NPAD * KPAD;
  for (int i = blockIdx.x * blockDim.x + threadIdx.x; i < total;
       i += gridDim.x * blockDim.x) {
    int n = i / KPAD, k = i % KPAD;
    float v = (n < Nsrc && k < Ksrc) ? src[k * Nsrc + n] : 0.0f;
    dst[i] = f2bf(v);
  }
}

__global__ void cvt_bf16(const float* __restrict__ src, u16* __restrict__ dst,
                         int total) {
  for (int i = blockIdx.x * blockDim.x + threadIdx.x; i < total;
       i += gridDim.x * blockDim.x)
    dst[i] = f2bf(src[i]);
}

// ------------------------------------------------- narrow-N WMMA GEMM -------
// A: bf16 [M,K] row-major.  BT: bf16 [NPAD,K] (col-major weights).
// Each wave: 16 rows x NPAD cols.  ACT: 0=none, 2=tanh.  Out fp32 [M,NOUT].
template <int K, int NPAD, int NOUT, int ACT>
__global__ __launch_bounds__(256) void gemm_narrow(const u16* __restrict__ A,
                                                   const u16* __restrict__ BT,
                                                   const float* __restrict__ bias,
                                                   float* __restrict__ outF) {
  constexpr int NF = NPAD / 16;
  const int tid = threadIdx.x, wid = tid >> 5, lane = tid & 31;
  const int lr = lane & 15, lhi = lane >> 4;
  const int row0 = (blockIdx.x * 8 + wid) * 16;
  const int r = row0 + lr;

  v8f acc[NF];
  for (int i = 0; i < NF; ++i) acc[i] = (v8f){0,0,0,0,0,0,0,0};

  for (int kk = 0; kk < K / 32; ++kk) {
    Frag16 a;
    const int kb = kk * 32 + lhi * 8;
    a.q[0] = *(const u32x4*)&A[(size_t)r * K + kb];
    a.q[1] = *(const u32x4*)&A[(size_t)r * K + kb + 16];
    const int kbase = kk * 32 + lhi * 8;
#pragma unroll
    for (int ni = 0; ni < NF; ++ni) {
      Frag16 b;
      const int n = ni * 16 + lr;
      b.q[0] = *(const u32x4*)&BT[(size_t)n * K + kbase];
      b.q[1] = *(const u32x4*)&BT[(size_t)n * K + kbase + 16];
      acc[ni] = wmma_bf16(a.v, b.v, acc[ni]);
    }
  }
#pragma unroll
  for (int ni = 0; ni < NF; ++ni) {
    const int n = ni * 16 + lr;
    if (n < NOUT) {
      const float bv = bias[n];
#pragma unroll
      for (int v = 0; v < 8; ++v) {
        const int m = row0 + v + 8 * lhi;
        float x = acc[ni][v] + bv;
        if (ACT == 2) x = tanhf(x);
        outF[(size_t)m * NOUT + n] = x;
      }
    }
  }
}

// ----------------------------------------------------------- dense2 ---------
__global__ void dense2(const float* __restrict__ f, const float* __restrict__ W2,
                       const float* __restrict__ b2, float* __restrict__ theta,
                       int B) {
  int b = blockIdx.x * blockDim.x + threadIdx.x;
  if (b >= B) return;
  const float* fr = f + b * 20;
#pragma unroll
  for (int j = 0; j < 6; ++j) {
    float s = b2[j];
#pragma unroll
    for (int i = 0; i < 20; ++i) s += fr[i] * W2[i * 6 + j];
    theta[b * 6 + j] = tanhf(s);
  }
}

// ----------------------------------------------------------- sampler --------
__global__ void sampler(const float* __restrict__ U, const float* __restrict__ theta,
                        u16* __restrict__ samp, int B) {
  int tid = blockIdx.x * blockDim.x + threadIdx.x;
  if (tid >= B * 1600) return;
  int b = tid / 1600, n = tid % 1600;
  int oy = n / 40, ox = n % 40;
  const float* th = theta + b * 6;
  float xt = -1.0f + ox * (2.0f / 39.0f);
  float yt = -1.0f + oy * (2.0f / 39.0f);
  float x = (th[0] * xt + th[1] * yt + th[2] + 1.0f) * 20.0f;
  float y = (th[3] * xt + th[4] * yt + th[5] + 1.0f) * 20.0f;
  float x0f = floorf(x), y0f = floorf(y);
  float x1f = x0f + 1.0f, y1f = y0f + 1.0f;
  int x0 = min(max((int)x0f, 0), 39), x1 = min(max((int)x1f, 0), 39);
  int y0 = min(max((int)y0f, 0), 39), y1 = min(max((int)y1f, 0), 39);
  const float* Ub = U + (size_t)b * 1600;
  float Ia = Ub[y0 * 40 + x0], Ibv = Ub[y1 * 40 + x0];
  float Ic = Ub[y0 * 40 + x1], Idv = Ub[y1 * 40 + x1];
  float wa = (x1f - x) * (y1f - y), wb = (x1f - x) * (y - y0f);
  float wc = (x - x0f) * (y1f - y), wd = (x - x0f) * (y - y0f);
  samp[tid] = f2bf(wa * Ia + wb * Ibv + wc * Ic + wd * Idv);
}

// --------------------------------- conv1 (implicit GEMM, K=25->32) ----------
__global__ __launch_bounds__(256) void conv1_wmma(const u16* __restrict__ samp,
                                                  const u16* __restrict__ BTc1,
                                                  const float* __restrict__ bias,
                                                  u16* __restrict__ out, int B) {
  const int tid = threadIdx.x, wid = tid >> 5, lane = tid & 31;
  const int lr = lane & 15, lhi = lane >> 4;
  const int p0 = (blockIdx.x * 8 + wid) * 16;        // 16 output pixels / wave
  const int p = p0 + lr;
  const int b = p / 400, rem = p % 400, oy = rem / 20, ox = rem % 20;

  Frag16 a;
#pragma unroll
  for (int e = 0; e < 16; ++e) {
    const int k = lhi * 8 + (e & 7) + ((e >> 3) << 4);
    u16 val = 0;
    if (k < 25) {
      const int ky = k / 5, kx = k % 5;
      const int iy = 2 * oy + ky - 2, ix = 2 * ox + kx - 2;
      if ((unsigned)iy < 40u && (unsigned)ix < 40u)
        val = samp[(size_t)b * 1600 + iy * 40 + ix];
    }
    a.u[e] = val;
  }
  Frag16 bf;
  const int kbase = lhi * 8;
  bf.q[0] = *(const u32x4*)&BTc1[lr * 32 + kbase];
  bf.q[1] = *(const u32x4*)&BTc1[lr * 32 + kbase + 16];

  v8f acc = (v8f){0,0,0,0,0,0,0,0};
  acc = wmma_bf16(a.v, bf.v, acc);

  const float bv = bias[lr];
#pragma unroll
  for (int v = 0; v < 8; ++v) {
    const int po = p0 + v + 8 * lhi;
    out[(size_t)po * 16 + lr] = f2bf(fmaxf(acc[v] + bv, 0.0f));
  }
}

// --------------------------------- conv2 (implicit GEMM, K=400->416) --------
__global__ __launch_bounds__(256) void conv2_wmma(const u16* __restrict__ in,
                                                  const u16* __restrict__ BTc2,
                                                  const float* __restrict__ bias,
                                                  u16* __restrict__ out, int B) {
  __shared__ __align__(16) u16 Bs[16 * 416];
  const int tid = threadIdx.x;
  for (int q = tid; q < (16 * 416) / 8; q += 256)    // 13.3 KB -> LDS
    ((u32x4*)Bs)[q] = ((const u32x4*)BTc2)[q];
  __syncthreads();

  const int wid = tid >> 5, lane = tid & 31;
  const int lr = lane & 15, lhi = lane >> 4;
  const int p0 = (blockIdx.x * 8 + wid) * 16;
  const int p = p0 + lr;
  const int b = p / 100, rem = p % 100, oy = rem / 10, ox = rem % 10;

  v8f acc = (v8f){0,0,0,0,0,0,0,0};
#pragma unroll 4
  for (int kk = 0; kk < 13; ++kk) {                  // 13 * 32 = 416
    Frag16 a;
#pragma unroll
    for (int c = 0; c < 2; ++c) {
      const int ks = kk * 32 + lhi * 8 + c * 16;     // 8 contiguous ci values
      u32x4 val = (u32x4){0, 0, 0, 0};
      if (ks < 400) {
        const int ky = ks / 80, r2 = ks % 80, kx = r2 / 16, ci = r2 % 16;
        const int iy = 2 * oy + ky - 2, ix = 2 * ox + kx - 2;
        if ((unsigned)iy < 20u && (unsigned)ix < 20u)
          val = *(const u32x4*)&in[((size_t)b * 400 + iy * 20 + ix) * 16 + ci];
      }
      a.q[c] = val;
    }
    Frag16 bf;
    const int kbase = kk * 32 + lhi * 8;
    bf.q[0] = *(const u32x4*)&Bs[lr * 416 + kbase];
    bf.q[1] = *(const u32x4*)&Bs[lr * 416 + kbase + 16];
    acc = wmma_bf16(a.v, bf.v, acc);
  }
  const float bv = bias[lr];
#pragma unroll
  for (int v = 0; v < 8; ++v) {
    const int po = p0 + v + 8 * lhi;
    out[(size_t)po * 16 + lr] = f2bf(fmaxf(acc[v] + bv, 0.0f));
  }
}

// ------------------------------ big GEMM3: 8192x1024, K=1600, relu ----------
// 128x128 block, BK=32; A/B tiles DMA'd into LDS by the Tensor Data Mover
// (wave 0 issues both descriptors, waits TENSORcnt, then workgroup barrier).
// 8 waves (4x2), each 32x64 -> 2x4 frags.
__global__ __launch_bounds__(256) void gemm_big(const u16* __restrict__ A,
                                                const u16* __restrict__ BT,
                                                const float* __restrict__ bias,
                                                u16* __restrict__ outBF,
                                                int M, int N, int K) {
  __shared__ __align__(16) u16 As[128 * 32];
  __shared__ __align__(16) u16 Bs[128 * 32];
  const int tid = threadIdx.x, wid = tid >> 5, lane = tid & 31;
  const int lr = lane & 15, lhi = lane >> 4;
  const int mw = wid & 3, nw = wid >> 2;
  const int m0 = blockIdx.x * 128, n0 = blockIdx.y * 128;
  const unsigned asOff = (unsigned)(uintptr_t)(void*)As;   // LDS byte offset
  const unsigned bsOff = (unsigned)(uintptr_t)(void*)Bs;

  v8f acc[2][4];
#pragma unroll
  for (int i = 0; i < 2; ++i)
#pragma unroll
    for (int j = 0; j < 4; ++j) acc[i][j] = (v8f){0,0,0,0,0,0,0,0};

  for (int k0 = 0; k0 < K; k0 += 32) {
    if (wid == 0) {                                  // TDM: 32x128 bf16 tiles
      tdm_load_2d(asOff, A + (size_t)m0 * K + k0,
                  (unsigned)K, (unsigned)M, (unsigned)K, 32u, 128u);
      tdm_load_2d(bsOff, BT + (size_t)n0 * K + k0,
                  (unsigned)K, (unsigned)N, (unsigned)K, 32u, 128u);
      __builtin_amdgcn_s_wait_tensorcnt(0);
    }
    __syncthreads();

    Frag16 a[2], b[4];
    const int kb = lhi * 8, kbase = lhi * 8;
#pragma unroll
    for (int mi = 0; mi < 2; ++mi) {
      const int r = mw * 32 + mi * 16 + lr;
      a[mi].q[0] = *(const u32x4*)&As[r * 32 + kb];
      a[mi].q[1] = *(const u32x4*)&As[r * 32 + kb + 16];
    }
#pragma unroll
    for (int ni = 0; ni < 4; ++ni) {
      const int n = nw * 64 + ni * 16 + lr;
      b[ni].q[0] = *(const u32x4*)&Bs[n * 32 + kbase];
      b[ni].q[1] = *(const u32x4*)&Bs[n * 32 + kbase + 16];
    }
#pragma unroll
    for (int mi = 0; mi < 2; ++mi)
#pragma unroll
      for (int ni = 0; ni < 4; ++ni)
        acc[mi][ni] = wmma_bf16(a[mi].v, b[ni].v, acc[mi][ni]);
    __syncthreads();
  }
#pragma unroll
  for (int mi = 0; mi < 2; ++mi)
#pragma unroll
    for (int ni = 0; ni < 4; ++ni) {
      const int n = n0 + nw * 64 + ni * 16 + lr;
      const float bv = bias[n];
#pragma unroll
      for (int v = 0; v < 8; ++v) {
        const int m = m0 + mw * 32 + mi * 16 + v + 8 * lhi;
        outBF[(size_t)m * N + n] = f2bf(fmaxf(acc[mi][ni][v] + bv, 0.0f));
      }
    }
}

// ---------------------------------------------------------------------------
extern "C" void kernel_launch(void* const* d_in, const int* in_sizes, int n_in,
                              void* d_out, int out_size, void* d_ws, size_t ws_size,
                              hipStream_t stream) {
  const float* img = (const float*)d_in[0];
  const float* W1  = (const float*)d_in[1];
  const float* b1  = (const float*)d_in[2];
  const float* W2  = (const float*)d_in[3];
  const float* b2  = (const float*)d_in[4];
  const float* K1  = (const float*)d_in[5];
  const float* bc1 = (const float*)d_in[6];
  const float* K2  = (const float*)d_in[7];
  const float* bc2 = (const float*)d_in[8];
  const float* W3  = (const float*)d_in[9];
  const float* b3  = (const float*)d_in[10];
  const float* W4  = (const float*)d_in[11];
  const float* b4  = (const float*)d_in[12];
  float* out = (float*)d_out;

  const int B = in_sizes[0] / 1600;                  // 8192

  char* ws = (char*)d_ws;
  size_t o = 0;
  auto take = [&](size_t bytes) {
    char* p = ws + o;
    o = (o + bytes + 255) & ~(size_t)255;
    return p;
  };
  u16*   imgBF  = (u16*)take((size_t)B * 1600 * 2);
  float* fbuf   = (float*)take((size_t)B * 20 * 4);
  float* theta  = (float*)take((size_t)B * 6 * 4);
  u16*   sampBF = (u16*)take((size_t)B * 1600 * 2);
  u16*   c1BF   = (u16*)take((size_t)B * 6400 * 2);
  u16*   c2BF   = (u16*)take((size_t)B * 1600 * 2);
  u16*   a3BF   = (u16*)take((size_t)B * 1024 * 2);
  u16*   BT1    = (u16*)take((size_t)32 * 1600 * 2);
  u16*   BT3    = (u16*)take((size_t)1024 * 1600 * 2);
  u16*   BT4    = (u16*)take((size_t)16 * 1024 * 2);
  u16*   BTc1   = (u16*)take((size_t)16 * 32 * 2);
  u16*   BTc2   = (u16*)take((size_t)16 * 416 * 2);
  (void)ws_size; (void)n_in; (void)out_size;

  auto cdiv = [](int a, int b) { return (a + b - 1) / b; };

  // ---- weight/activation prep (bf16, transposed, padded) ----
  cvt_bf16<<<cdiv(B * 1600, 256), 256, 0, stream>>>(img, imgBF, B * 1600);
  prep_bt<<<cdiv(32 * 1600, 256), 256, 0, stream>>>(W1, BT1, 20, 32, 1600, 1600);
  prep_bt<<<cdiv(1024 * 1600, 256), 256, 0, stream>>>(W3, BT3, 1024, 1024, 1600, 1600);
  prep_bt<<<cdiv(16 * 1024, 256), 256, 0, stream>>>(W4, BT4, 10, 16, 1024, 1024);
  prep_bt<<<cdiv(16 * 32, 256), 256, 0, stream>>>(K1, BTc1, 16, 16, 25, 32);
  prep_bt<<<cdiv(16 * 416, 256), 256, 0, stream>>>(K2, BTc2, 16, 16, 400, 416);

  // ---- localization: tanh(img @ W1 + b1) -> tanh(f @ W2 + b2) ----
  gemm_narrow<1600, 32, 20, 2><<<B / 128, 256, 0, stream>>>(imgBF, BT1, b1, fbuf);
  dense2<<<cdiv(B, 256), 256, 0, stream>>>(fbuf, W2, b2, theta, B);

  // ---- spatial transformer (bilinear sample) ----
  sampler<<<cdiv(B * 1600, 256), 256, 0, stream>>>(img, theta, sampBF, B);

  // ---- classifier ----
  conv1_wmma<<<(B * 25) / 8, 256, 0, stream>>>(sampBF, BTc1, bc1, c1BF, B);
  conv2_wmma<<<(B * 100 / 16) / 8, 256, 0, stream>>>(c1BF, BTc2, bc2, c2BF, B);
  gemm_big<<<dim3(B / 128, 1024 / 128), 256, 0, stream>>>(c2BF, BT3, b3, a3BF,
                                                          B, 1024, 1600);
  gemm_narrow<1024, 16, 10, 0><<<B / 128, 256, 0, stream>>>(a3BF, BT4, b4, out);
}
